// GlobalAttention_28089086116736
// MI455X (gfx1250) — matrix operand order repaired and checked by reading, not compiled, not run on hardware
//
#include <hip/hip_runtime.h>
#include <hip/hip_bf16.h>

typedef __attribute__((ext_vector_type(16))) __bf16 v16bf;
typedef __attribute__((ext_vector_type(8)))  __bf16 v8bf;
typedef __attribute__((ext_vector_type(8)))  float  v8f;

#define NEG_INF (-1e9f)

constexpr int Bsz = 8, Tlen = 2048, Slen = 4096, Dm = 128;

// Assemble a 16-element bf16 WMMA fragment from two aligned 8-element chunks.
__device__ __forceinline__ v16bf frag2(const __bf16* p0, const __bf16* p1) {
    v8bf a = *(const v8bf*)p0;
    v8bf b = *(const v8bf*)p1;
    v16bf f;
#pragma unroll
    for (int i = 0; i < 8; ++i) { f[i] = a[i]; f[i + 8] = b[i]; }
    return f;
}

__device__ __forceinline__ v8f wmma_bf16(v16bf a, v16bf b, v8f c) {
    // (neg_a, A, neg_b, B, c_mod, C, reuse_a, reuse_b)
    return __builtin_amdgcn_wmma_f32_16x16x32_bf16(false, a, false, b, (short)0, c,
                                                   false, false);
}

// ---------------- prep: f32 -> bf16 staging ----------------
__global__ void cvt_bf16_kernel(const float* __restrict__ in, __bf16* __restrict__ out, int n) {
    int i = blockIdx.x * blockDim.x + threadIdx.x;
    if (i < n) out[i] = (__bf16)in[i];
}

// LDS-tiled transpose: mbT[b][d][s] = mb[b][s][d], coalesced both sides.
__global__ __launch_bounds__(256) void mbT_kernel(const float* __restrict__ in,
                                                  __bf16* __restrict__ out) {
    __shared__ float tile[64][65];           // 65 pitch -> conflict-free columns
    const int tilesPerB = (Slen / 64) * (Dm / 64);   // 128
    const int b  = blockIdx.x / tilesPerB;
    const int r  = blockIdx.x % tilesPerB;
    const int s0 = (r % (Slen / 64)) * 64;
    const int d0 = (r / (Slen / 64)) * 64;
    const int c  = threadIdx.x & 63, q = threadIdx.x >> 6;   // q: 0..3
#pragma unroll
    for (int ss = 0; ss < 64; ss += 4)
        tile[ss + q][c] = in[((size_t)b * Slen + s0 + ss + q) * Dm + d0 + c];
    __syncthreads();
#pragma unroll
    for (int dd = 0; dd < 64; dd += 4)
        out[((size_t)b * Dm + d0 + dd + q) * Slen + s0 + c] = (__bf16)tile[c][dd + q];
}

// ---------------- fused attention kernel ----------------
// Grid: B*T/16 workgroups, 256 threads (8 wave32s). Each workgroup owns a
// 16-row strip of T and the full S=4096 score row in LDS (bf16, 128KB).
__global__ __launch_bounds__(256) void attn_fused_kernel(
    const __bf16* __restrict__ qbf,   // [B,T,128]
    const __bf16* __restrict__ mbbf,  // [B,S,128]
    const __bf16* __restrict__ mbT,   // [B,128,S]
    const __bf16* __restrict__ wibf,  // [128,128]
    const __bf16* __restrict__ wobf,  // [128,256]
    const long long* __restrict__ lens,
    float* __restrict__ out_attn,     // [B,T,128]
    float* __restrict__ out_probs,    // [B,T,S]
    float* __restrict__ out_align)    // [B,T,S]
{
    extern __shared__ __align__(16) char smem[];
    __bf16* P_sm = (__bf16*)smem;            // 16*4096 bf16 = 128KB strip
    __bf16* h_sm = P_sm + 16 * Slen;         // 16*128
    __bf16* c_sm = h_sm + 16 * Dm;           // 16*128
    float*  red  = (float*)(c_sm + 16 * Dm); // 16*16
    float*  maxv = red + 256;                // 16
    float*  invv = maxv + 16;                // 16

    const int wg   = blockIdx.x;
    const int b    = wg / (Tlen / 16);
    const int t0   = (wg % (Tlen / 16)) * 16;
    const int tid  = threadIdx.x;
    const int wv   = tid >> 5;   // wave 0..7
    const int lane = tid & 31;
    const int hi   = lane >> 4;
    const int ln   = lane & 15;
    const long long len = lens[b];

    // ---- phase 1: h = query @ W_in^T  (16x128, wave wv owns cols 16wv..) ----
    {
        const __bf16* qrow = qbf + (size_t)(b * Tlen + t0 + ln) * Dm;
        const __bf16* wrow = wibf + (size_t)(16 * wv + ln) * Dm;
        v8f acc = {};
#pragma unroll
        for (int j = 0; j < 4; ++j) {
            int k0 = 32 * j;
            v16bf a  = frag2(qrow + k0 + 8 * hi, qrow + k0 + 16 + 8 * hi);
            v16bf bm = frag2(wrow + k0 + 8 * hi, wrow + k0 + 16 + 8 * hi);
            acc = wmma_bf16(a, bm, acc);
        }
#pragma unroll
        for (int i = 0; i < 8; ++i)
            h_sm[(i + 8 * hi) * Dm + 16 * wv + ln] = (__bf16)acc[i];
    }
    __syncthreads();

    // ---- phase 2: align = h @ mb^T, mask, write raw f32, stash bf16 in LDS ----
    v16bf afr[4];
    {
        const __bf16* hrow = h_sm + ln * Dm;
#pragma unroll
        for (int j = 0; j < 4; ++j)
            afr[j] = frag2(hrow + 32 * j + 8 * hi, hrow + 32 * j + 16 + 8 * hi);
    }
    for (int st = wv; st < Slen / 16; st += 8) {
        const int s0 = st * 16;
        const __bf16* mrow = mbbf + ((size_t)b * Slen + s0 + ln) * Dm;
        v8f acc = {};
#pragma unroll
        for (int j = 0; j < 4; ++j) {
            v16bf bm = frag2(mrow + 32 * j + 8 * hi, mrow + 32 * j + 16 + 8 * hi);
            acc = wmma_bf16(afr[j], bm, acc);
        }
        const bool valid = (long long)(s0 + ln) < len;
#pragma unroll
        for (int i = 0; i < 8; ++i) {
            const int m = i + 8 * hi;
            const float sc = valid ? acc[i] : NEG_INF;
            out_align[((size_t)(b * Tlen + t0 + m)) * Slen + (s0 + ln)] = sc;
            P_sm[m * Slen + s0 + ln] = (__bf16)sc;
        }
    }
    __syncthreads();

    // ---- phase 3: row max (vectorized), then e=exp(x-max) written in place
    //              while accumulating the row sum ----
    const int r = tid >> 4, chunk = tid & 15;
    {
        __bf16* prow = P_sm + r * Slen + chunk * 256;   // exclusive 256-elem region
        float mx = -3.0e38f;
        for (int s = 0; s < 256; s += 8) {
            v8bf v = *(const v8bf*)(prow + s);
#pragma unroll
            for (int e = 0; e < 8; ++e) mx = fmaxf(mx, (float)v[e]);
        }
        red[r * 16 + chunk] = mx;
        __syncthreads();
        if (tid < 16) {
            float mm = red[tid * 16];
#pragma unroll
            for (int c = 1; c < 16; ++c) mm = fmaxf(mm, red[tid * 16 + c]);
            maxv[tid] = mm;
        }
        __syncthreads();
        const float mv = maxv[r];
        float sm = 0.f;
        for (int s = 0; s < 256; s += 8) {
            v8bf v = *(const v8bf*)(prow + s);
            v8bf ev;
#pragma unroll
            for (int e = 0; e < 8; ++e) {
                float x = __expf((float)v[e] - mv);
                sm += x;
                ev[e] = (__bf16)x;
            }
            *(v8bf*)(prow + s) = ev;                    // P_sm now holds e
        }
        red[r * 16 + chunk] = sm;
        __syncthreads();
        if (tid < 16) {
            float ss = 0.f;
#pragma unroll
            for (int c = 0; c < 16; ++c) ss += red[tid * 16 + c];
            invv[tid] = 1.0f / ss;
        }
        __syncthreads();
    }

    // ---- phase 4: align_vectors = e * inv  (vector loads + float4 stores;
    //              no exp, no LDS writeback — inv folded into context) ----
    for (int idx = tid * 8; idx < 16 * Slen; idx += 256 * 8) {
        const int rr = idx >> 12, s = idx & (Slen - 1);
        const v8bf e = *(const v8bf*)(P_sm + idx);
        const float inv = invv[rr];
        float4 p0 = make_float4((float)e[0] * inv, (float)e[1] * inv,
                                (float)e[2] * inv, (float)e[3] * inv);
        float4 p1 = make_float4((float)e[4] * inv, (float)e[5] * inv,
                                (float)e[6] * inv, (float)e[7] * inv);
        float* dst = out_probs + ((size_t)(b * Tlen + t0 + rr)) * Slen + s;
        *(float4*)dst       = p0;
        *((float4*)dst + 1) = p1;
    }
    // no barrier needed: phase 5 reads P_sm/invv (final) and writes only c_sm

    // ---- phase 5: context c = inv[m] * (e @ mb)  (K=4096, out of LDS) ----
    {
        const __bf16* prow  = P_sm + ln * Slen;
        const __bf16* mtrow = mbT + ((size_t)b * Dm + 16 * wv + ln) * Slen;
        v8f acc = {};
        for (int j = 0; j < Slen / 32; ++j) {
            const int k0 = 32 * j;
            v16bf a  = frag2(prow + k0 + 8 * hi, prow + k0 + 16 + 8 * hi);
            v16bf bm = frag2(mtrow + k0 + 8 * hi, mtrow + k0 + 16 + 8 * hi);
            acc = wmma_bf16(a, bm, acc);
        }
#pragma unroll
        for (int i = 0; i < 8; ++i)
            c_sm[(i + 8 * hi) * Dm + 16 * wv + ln] = (__bf16)(acc[i] * invv[i + 8 * hi]);
    }
    __syncthreads();

    // ---- phase 6: attn_h = tanh(concat(c, q) @ W_out^T), K=256 ----
    {
        const __bf16* crow = c_sm + ln * Dm;
        const __bf16* qrow = qbf + (size_t)(b * Tlen + t0 + ln) * Dm;
        const __bf16* wrow = wobf + (size_t)(16 * wv + ln) * 256;
        v8f acc = {};
#pragma unroll
        for (int j = 0; j < 8; ++j) {
            const int k0 = 32 * j;
            v16bf a = (k0 < 128)
                ? frag2(crow + k0 + 8 * hi, crow + k0 + 16 + 8 * hi)
                : frag2(qrow + (k0 - 128) + 8 * hi, qrow + (k0 - 128) + 16 + 8 * hi);
            v16bf bm = frag2(wrow + k0 + 8 * hi, wrow + k0 + 16 + 8 * hi);
            acc = wmma_bf16(a, bm, acc);
        }
#pragma unroll
        for (int i = 0; i < 8; ++i)
            out_attn[((size_t)(b * Tlen + t0 + i + 8 * hi)) * Dm + 16 * wv + ln] =
                tanhf(acc[i]);
    }
}

// ---------------- host launch ----------------
extern "C" void kernel_launch(void* const* d_in, const int* in_sizes, int n_in,
                              void* d_out, int out_size, void* d_ws, size_t ws_size,
                              hipStream_t stream) {
    (void)in_sizes; (void)n_in; (void)out_size; (void)ws_size;

    const float*     query = (const float*)d_in[0];   // [B,T,128]
    const float*     mb    = (const float*)d_in[1];   // [B,S,128]
    const long long* lens  = (const long long*)d_in[2];
    const float*     W_in  = (const float*)d_in[3];   // [128,128]
    const float*     W_out = (const float*)d_in[4];   // [128,256]

    float* out_attn  = (float*)d_out;                               // B*T*128
    float* out_probs = out_attn + (size_t)Bsz * Tlen * Dm;          // B*T*S
    float* out_align = out_probs + (size_t)Bsz * Tlen * Slen;       // B*T*S

    // workspace: bf16 staging buffers (~21 MB total)
    __bf16* qbf  = (__bf16*)d_ws;                       // B*T*128
    __bf16* mbbf = qbf  + (size_t)Bsz * Tlen * Dm;      // B*S*128
    __bf16* mbT  = mbbf + (size_t)Bsz * Slen * Dm;      // B*128*S
    __bf16* wibf = mbT  + (size_t)Bsz * Slen * Dm;      // 128*128
    __bf16* wobf = wibf + 128 * 128;                    // 128*256

    const int nq = Bsz * Tlen * Dm;      // 2,097,152
    const int nm = Bsz * Slen * Dm;      // 4,194,304
    cvt_bf16_kernel<<<(nq + 255) / 256, 256, 0, stream>>>(query, qbf, nq);
    cvt_bf16_kernel<<<(nm + 255) / 256, 256, 0, stream>>>(mb, mbbf, nm);
    cvt_bf16_kernel<<<(128 * 128 + 255) / 256, 256, 0, stream>>>(W_in, wibf, 128 * 128);
    cvt_bf16_kernel<<<(128 * 256 + 255) / 256, 256, 0, stream>>>(W_out, wobf, 128 * 256);
    mbT_kernel<<<Bsz * (Slen / 64) * (Dm / 64), 256, 0, stream>>>(mb, mbT);

    const size_t smem = (size_t)16 * Slen * sizeof(__bf16)   // P_sm 128KB
                      + (size_t)2 * 16 * Dm * sizeof(__bf16) // h_sm + c_sm
                      + (256 + 32) * sizeof(float);          // reductions
    (void)hipFuncSetAttribute((const void*)attn_fused_kernel,
                              hipFuncAttributeMaxDynamicSharedMemorySize, (int)smem);

    const int grid = Bsz * (Tlen / 16); // 1024 workgroups
    attn_fused_kernel<<<grid, 256, smem, stream>>>(qbf, mbbf, mbT, wibf, wobf, lens,
                                                   out_attn, out_probs, out_align);
}
